// MotionRNN_cell_37589553774668
// MI455X (gfx1250) — matrix operand order repaired and checked by reading, not compiled, not run on hardware
//
#include <hip/hip_runtime.h>
#include <hip/hip_bf16.h>

// ---------------------------------------------------------------------------
// Types / helpers
// ---------------------------------------------------------------------------
typedef __bf16 bf16;
typedef __attribute__((ext_vector_type(16))) __bf16 v16bf;
typedef __attribute__((ext_vector_type(8)))  __bf16 v8bf;
typedef __attribute__((ext_vector_type(8)))  float  v8f;
typedef int v4i __attribute__((vector_size(16)));

#define CIN   64
#define HW    128
#define CHW   (64 * 128 * 128)
#define FULL  (4 * CHW)
#define HP    4096
#define H18   (4 * 18 * HP)
#define H16   (4 * 16 * HP)
#define H9    (4 * 9  * HP)
#define WARPN (4 * 16 * HP * 9)

// CDNA5 async global->LDS path (guarded; sync fallback keeps compile safe)
#if defined(__AMDGCN__) && __has_builtin(__builtin_amdgcn_global_load_async_to_lds_b128)
#define HAVE_ASYNC 1
#define ASYNC_B128(gsrc, ldst)                                                \
    __builtin_amdgcn_global_load_async_to_lds_b128(                           \
        (__attribute__((address_space(1))) v4i*)(gsrc),                       \
        (__attribute__((address_space(3))) v4i*)(ldst), 0, 0)
#if __has_builtin(__builtin_amdgcn_s_wait_asynccnt)
#define WAIT_ASYNC() __builtin_amdgcn_s_wait_asynccnt(0)
#else
#define WAIT_ASYNC() asm volatile("s_wait_asynccnt 0" ::: "memory")
#endif
#endif

__device__ __forceinline__ float sigm(float x) { return 1.f / (1.f + __expf(-x)); }

__device__ __forceinline__ bf16 f2bf(float f) {
    union { float f; unsigned u; } a; a.f = f;
    unsigned r = (a.u + 0x7FFFu + ((a.u >> 16) & 1u)) >> 16;
    union { unsigned short s; bf16 b; } o; o.s = (unsigned short)r;
    return o.b;
}
__device__ __forceinline__ float bf2f(bf16 b) {
    union { unsigned short s; bf16 b; } i; i.b = b;
    union { unsigned u; float f; } o; o.u = (unsigned)i.s << 16;
    return o.f;
}

// ---------------------------------------------------------------------------
// WMMA implicit-GEMM 3x3 conv (bf16 in, f32 out), dual input.
// in  : bf16 (B,64,128,128)
// wpk : bf16 packed [cob16][kh][co16][kpos][ci32] (4608 halves per (cob,kh))
// grid = (128 rows, OC/16, B), block 256 (8 waves, 16 px each).
// ---------------------------------------------------------------------------
__global__ __launch_bounds__(256) void conv3x3_dual_wmma(
    const bf16* __restrict__ inA, const bf16* __restrict__ inB,
    const bf16* __restrict__ wA,  const bf16* __restrict__ wB,
    const float* __restrict__ biasA, const float* __restrict__ biasB,
    float* __restrict__ out, int OC)
{
    const int y = blockIdx.x, cob = blockIdx.y, b = blockIdx.z;
    const int tid = threadIdx.x, lane = tid & 31;
    const int wave = tid >> 5, x0 = wave * 16;
    const int n = lane & 15, hk = lane >> 4;

    __shared__ bf16 s_act[3 * 132 * 32];   // [row][col(132)][ci32]
    __shared__ bf16 s_w[16 * 9 * 32];      // [co][kpos][ci32]

    // zero halo columns 0 and 129 (never rewritten afterwards)
    if (tid < 192) {
        int ci = tid & 31, t = tid >> 5;         // t in 0..5
        int r = t >> 1, border = t & 1;
        s_act[(r * 132 + (border ? 129 : 0)) * 32 + ci] = f2bf(0.f);
    }

    v8f acc = {};
#pragma unroll
    for (int r = 0; r < 8; ++r) {
        int co = cob * 16 + r + 8 * hk;
        float bv = biasA ? biasA[co] : 0.f;
        if (biasB) bv += biasB[co];
        acc[r] = bv;
    }

    for (int src = 0; src < 2; ++src) {
        const bf16* in  = src ? inB : inA;
        const bf16* wpk = src ? wB : wA;
        if (!in) break;
        for (int kh = 0; kh < 2; ++kh) {
            // ---- weights: contiguous 4608 halves, async copy to LDS ----
            const bf16* wsrc = wpk + ((size_t)(cob * 2 + kh)) * 4608;
#ifdef HAVE_ASYNC
            for (int i = tid; i < 576; i += 256)
                ASYNC_B128(wsrc + i * 8, s_w + i * 8);
#else
            for (int i = tid; i < 576; i += 256)
                *(v8bf*)(s_w + i * 8) = *(const v8bf*)(wsrc + i * 8);
#endif
            if (kh == 0 && tid < 192)   // prefetch second ci-half
                __builtin_prefetch(in + (((size_t)b * CIN + 32 + (tid & 31)) * HW + y) * HW, 0, 1);
            // ---- activations: 192 register-transposed 8x8 micro-tiles ----
            if (tid < 192) {
                int q = tid & 3, g = (tid >> 2) & 15, r = tid >> 6;
                int yy = y + r - 1;
                bf16* dst0 = &s_act[(r * 132 + 8 * g + 1) * 32 + q * 8];
                if ((unsigned)yy < HW) {
                    const bf16* sp = in + (((size_t)b * CIN + (kh * 32 + q * 8)) * HW + yy) * HW + 8 * g;
                    v8bf av[8];
#pragma unroll
                    for (int k = 0; k < 8; ++k)
                        av[k] = *(const v8bf*)(sp + (size_t)k * HW * HW);
#pragma unroll
                    for (int j = 0; j < 8; ++j) {
                        v8bf t;
#pragma unroll
                        for (int k = 0; k < 8; ++k) t[k] = av[k][j];
                        *(v8bf*)(dst0 + (size_t)j * 32) = t;
                    }
                } else {
                    v8bf z = {};
#pragma unroll
                    for (int j = 0; j < 8; ++j) *(v8bf*)(dst0 + (size_t)j * 32) = z;
                }
            }
#ifdef HAVE_ASYNC
            WAIT_ASYNC();
#endif
            __syncthreads();
            // ---- 9 shifted K=32 GEMMs ----
#pragma unroll
            for (int kp = 0; kp < 9; ++kp) {
                const int ky = kp / 3, kx = kp % 3;   // constants after unroll
                const bf16* bp = &s_act[((ky * 132) + (x0 + n + kx)) * 32 + 16 * hk];
                v16bf bfr = *(const v16bf*)bp;
                const bf16* wp = &s_w[(n * 9 + kp) * 32];
                v8bf alo = *(const v8bf*)(wp + 16 * hk);
                v8bf ahi = *(const v8bf*)(wp + 16 * hk + 8);
                v16bf afr = __builtin_shufflevector(alo, ahi,
                    0, 1, 2, 3, 4, 5, 6, 7, 8, 9, 10, 11, 12, 13, 14, 15);
                acc = __builtin_amdgcn_wmma_f32_16x16x32_bf16(
                    false, afr, false, bfr, (short)0, acc, false, false);
            }
            __syncthreads();
        }
    }
#pragma unroll
    for (int r = 0; r < 8; ++r) {
        int co = cob * 16 + r + 8 * hk;
        out[(((size_t)b * OC + co) * HW + y) * HW + (x0 + n)] = acc[r];
    }
}

// ---------------------------------------------------------------------------
// WMMA 1x1 conv (bf16 in), K=128 from two concatenated 64-ch inputs.
// w: bf16 [OC][128] (natural layout). grid=(P/128, OC/16, B), block 256.
// ---------------------------------------------------------------------------
__global__ __launch_bounds__(256) void conv1x1_dual_wmma(
    const bf16* __restrict__ inA, const bf16* __restrict__ inB,
    const bf16* __restrict__ w, const float* __restrict__ bias,
    float* __restrict__ out, int P, int OC)
{
    __shared__ bf16 s_in[128 * 128];   // [px][ci]
    __shared__ bf16 s_w[16 * 128];
    const int tid = threadIdx.x, lane = tid & 31, wave = tid >> 5;
    const int p0 = blockIdx.x * 128, cob = blockIdx.y, b = blockIdx.z;
    const int n = lane & 15, hk = lane >> 4;

    const bf16* wsrc = w + (size_t)cob * 16 * 128;   // 2048 halves = 256 b128
#ifdef HAVE_ASYNC
    ASYNC_B128(wsrc + tid * 8, s_w + tid * 8);
#else
    *(v8bf*)(s_w + tid * 8) = *(const v8bf*)(wsrc + tid * 8);
#endif
    {   // 256 transposed 8x8 micro-tiles
        int pg = tid & 15, cg = tid >> 4;            // px-group, ci-group
        const bf16* base = (cg < 8) ? inA : inB;
        int c0 = (cg & 7) * 8;
        v8bf av[8];
#pragma unroll
        for (int k = 0; k < 8; ++k)
            av[k] = *(const v8bf*)(base + ((size_t)b * 64 + c0 + k) * P + p0 + pg * 8);
#pragma unroll
        for (int j = 0; j < 8; ++j) {
            v8bf t;
#pragma unroll
            for (int k = 0; k < 8; ++k) t[k] = av[k][j];
            *(v8bf*)(&s_in[(pg * 8 + j) * 128 + cg * 8]) = t;
        }
    }
#ifdef HAVE_ASYNC
    WAIT_ASYNC();
#endif
    __syncthreads();

    v8f acc = {};
#pragma unroll
    for (int r = 0; r < 8; ++r) acc[r] = bias ? bias[cob * 16 + r + 8 * hk] : 0.f;

    const int px = wave * 16 + n;
#pragma unroll
    for (int kc = 0; kc < 4; ++kc) {
        const bf16* bp = &s_in[px * 128 + kc * 32 + 16 * hk];
        v16bf bfr = *(const v16bf*)bp;
        const bf16* wp = &s_w[n * 128 + kc * 32];
        v8bf alo = *(const v8bf*)(wp + 16 * hk);
        v8bf ahi = *(const v8bf*)(wp + 16 * hk + 8);
        v16bf afr = __builtin_shufflevector(alo, ahi,
            0, 1, 2, 3, 4, 5, 6, 7, 8, 9, 10, 11, 12, 13, 14, 15);
        acc = __builtin_amdgcn_wmma_f32_16x16x32_bf16(
            false, afr, false, bfr, (short)0, acc, false, false);
    }
#pragma unroll
    for (int r = 0; r < 8; ++r)
        out[((size_t)b * OC + cob * 16 + r + 8 * hk) * P + p0 + wave * 16 + n] = acc[r];
}

// ---------------------------------------------------------------------------
// Weight prep
// ---------------------------------------------------------------------------
// OIHW f32 -> packed bf16 [cob16][kh][co16][kp][ci32]
__global__ void pack_w3(const float* __restrict__ w, bf16* __restrict__ o, int OC) {
    int i = blockIdx.x * 256 + threadIdx.x;
    if (i >= OC * 576) return;
    int c32 = i & 31;
    int t = i >> 5;
    int kp = t % 9;
    int u = t / 9;
    int co = u & 15;
    int v = u >> 4;
    int kh = v & 1;
    int cb = v >> 1;
    o[i] = f2bf(w[((size_t)(cb * 16 + co) * 64 + kh * 32 + c32) * 9 + kp]);
}

__global__ void cvt_bf(const float* __restrict__ a, bf16* __restrict__ o, int N) {
    int i = blockIdx.x * 256 + threadIdx.x;
    if (i < N) o[i] = f2bf(a[i]);
}

// ---------------------------------------------------------------------------
// Elementwise kernels (fp32 math; optional bf16 shadow copies for conv inputs)
// ---------------------------------------------------------------------------
__global__ void ew_sub_bf(const float* a, const float* b, bf16* o) {
    size_t i = (size_t)blockIdx.x * 256 + threadIdx.x;
    if (i < FULL) o[i] = f2bf(a[i] - b[i]);
}

__global__ void gate_ifg(const float* A, const float* state, float* next, bf16* nextb, int nq) {
    size_t i = (size_t)blockIdx.x * 256 + threadIdx.x;
    if (i >= FULL) return;
    size_t b = i / CHW, r = i % CHW;
    const float* Ab = A + b * (size_t)nq * CHW;
    float v = sigm(Ab[CHW + r]) * state[i] + sigm(Ab[r]) * tanhf(Ab[2 * (size_t)CHW + r]);
    next[i] = v;
    if (nextb) nextb[i] = f2bf(v);
}

__global__ void ew_dif(const float* onc, const float* nn, bf16* dif) {
    size_t i = (size_t)blockIdx.x * 256 + threadIdx.x;
    if (i >= FULL) return;
    dif[i] = f2bf(sigm(onc[i]) * tanhf(nn[i]));
}

__global__ void ew_T(const float* A2, const float* s2o, const float* ns, float* T) {
    size_t i = (size_t)blockIdx.x * 256 + threadIdx.x;
    if (i >= FULL) return;
    size_t b = i / CHW, r = i % CHW;
    float osv = sigm(A2[b * (size_t)4 * CHW + 3 * (size_t)CHW + r] + s2o[i]);
    T[i] = osv * tanhf(ns[i]);
}

__global__ void ew_nc(const float* A3, const float* T, float* nc, bf16* ncb) {
    size_t i = (size_t)blockIdx.x * 256 + threadIdx.x;
    if (i >= FULL) return;
    size_t b = i / CHW, r = i % CHW;
    const float* Ab = A3 + b * (size_t)4 * CHW;
    float v = T[i] + sigm(Ab[r]) * tanhf(Ab[2 * (size_t)CHW + r]);
    nc[i] = v;
    ncb[i] = f2bf(v);
}

__global__ void ew_o(const float* A3, const float* oc2, float* o) {
    size_t i = (size_t)blockIdx.x * 256 + threadIdx.x;
    if (i >= FULL) return;
    size_t b = i / CHW, r = i % CHW;
    o[i] = sigm(A3[b * (size_t)4 * CHW + 3 * (size_t)CHW + r] + oc2[i]);
}

__global__ void ew_hh(const float* o, const float* cm, float* hh, bf16* hhb) {
    size_t i = (size_t)blockIdx.x * 256 + threadIdx.x;
    if (i >= FULL) return;
    float v = o[i] * tanhf(cm[i]);
    hh[i] = v;
    hhb[i] = f2bf(v);
}

__global__ void ew_gru(const float* u, const float* z, const float* Fl,
                       const float* Dl, float* nF, float* nD) {
    int i = blockIdx.x * 256 + threadIdx.x;
    if (i >= H18) return;
    float uv = u[i], Ft = uv * z[i] + (1.f - uv) * Fl[i];
    float nd = Dl[i] + 0.5f * (Fl[i] - Dl[i]);
    nD[i] = nd;
    nF[i] = Ft + nd;
}

__global__ void ew_final(const float* G, const float* Hh, const float* Hd,
                         const float* o, const float* x, float* nh) {
    size_t i = (size_t)blockIdx.x * 256 + threadIdx.x;
    if (i >= FULL) return;
    float g = sigm(G[i]);
    nh[i] = g * Hh[i] + (1.f - g) * Hd[i] + (1.f - o[i]) * x[i];
}

// ---------------------------------------------------------------------------
// Small half-res kernels (VALU; negligible FLOPs)
// ---------------------------------------------------------------------------
__global__ void conv1x1_naive(const float* inA, int CA, const float* inB, int CB,
                              const float* mulB, const float* w, const float* bias,
                              float* out, int OCc, int P, int act) {
    int i = blockIdx.x * 256 + threadIdx.x;
    if (i >= 4 * OCc * P) return;
    int p = i & (P - 1), t = i / P, oc = t % OCc, b = t / OCc;
    const float* wr = w + (size_t)oc * (CA + CB);
    float acc = bias ? bias[oc] : 0.f;
    for (int c = 0; c < CA; ++c)
        acc += inA[((size_t)b * CA + c) * P + p] * wr[c];
    if (inB)
        for (int c = 0; c < CB; ++c) {
            float v = inB[((size_t)b * CB + c) * P + p];
            if (mulB) v *= mulB[((size_t)b * CB + c) * P + p];
            acc += v * wr[CA + c];
        }
    if (act == 1) acc = sigm(acc);
    else if (act == 2) acc = tanhf(acc);
    out[i] = acc;
}

__global__ void enc_conv(const bf16* in, const float* w, const float* bias, float* out) {
    int i = blockIdx.x * 256 + threadIdx.x;
    if (i >= 4 * 16 * HP) return;
    int xo = i & 63, yo = (i >> 6) & 63, ce = (i >> 12) & 15, b = i >> 16;
    float acc = bias[ce];
    for (int ci = 0; ci < 64; ++ci) {
        const bf16* ip = in + (((size_t)b * 64 + ci) * HW + 2 * yo) * HW + 2 * xo;
        const float* wp = w + ((size_t)ce * 64 + ci) * 4;
        acc += bf2f(ip[0]) * wp[0] + bf2f(ip[1]) * wp[1]
             + bf2f(ip[HW]) * wp[2] + bf2f(ip[HW + 1]) * wp[3];
    }
    out[i] = acc;
}

__global__ void warp_kernel(const float* Henc, const float* NF, const float* mm, float* Hw) {
    int i = blockIdx.x * 256 + threadIdx.x;
    if (i >= 4 * 16 * HP) return;
    int x = i & 63, y = (i >> 6) & 63, ce = (i >> 12) & 15, b = i >> 16;
    int p = y * 64 + x;
    const float* img = Henc + ((size_t)b * 16 + ce) * HP;
    float* op = Hw + (size_t)i * 9;
#pragma unroll 1
    for (int k = 0; k < 9; ++k) {
        int g0 = k * 8 + b * 2;       // flat (b*18+c) channel index (torch reshape)
        float fx = NF[(size_t)g0 * HP + p];
        float fy = NF[(size_t)(g0 + 1) * HP + p];
        float gx = (float)x + fx, gy = (float)y + fy;
        float fgx = floorf(gx), fgy = floorf(gy);
        int x0i = (int)fgx, y0i = (int)fgy;
        float frx = gx - fgx, fry = gy - fgy;
        float acc = 0.f;
#pragma unroll
        for (int dy = 0; dy < 2; ++dy)
#pragma unroll
            for (int dx = 0; dx < 2; ++dx) {
                int xi = x0i + dx, yi = y0i + dy;
                if ((unsigned)xi < 64u && (unsigned)yi < 64u) {
                    float wgt = (dy ? fry : 1.f - fry) * (dx ? frx : 1.f - frx);
                    acc += wgt * img[yi * 64 + xi];
                }
            }
        op[k] = acc * mm[((size_t)b * 9 + k) * HP + p];
    }
}

__global__ void deconv_kernel(const float* in, const float* w, const float* bias,
                              float* out, bf16* outb) {
    size_t i = (size_t)blockIdx.x * 256 + threadIdx.x;
    if (i >= FULL) return;
    int X = i & (HW - 1), Y = (i >> 7) & (HW - 1), co = (i >> 14) & 63, b = i >> 20;
    float acc = bias[co];
#pragma unroll
    for (int ky = 0; ky < 4; ++ky) {
        int py = Y + ky - 2;
        if (py < 0 || py > 126 || (py & 1)) continue;
        int iy = py >> 1;
#pragma unroll
        for (int kx = 0; kx < 4; ++kx) {
            int px = X + kx - 2;
            if (px < 0 || px > 126 || (px & 1)) continue;
            int ix = px >> 1;
            for (int ci = 0; ci < 16; ++ci)
                acc += in[(((size_t)b * 16 + ci) * 64 + iy) * 64 + ix] *
                       w[(((size_t)co * 16 + ci) * 4 + ky) * 4 + kx];
        }
    }
    out[i] = acc;
    outb[i] = f2bf(acc);
}

// ---------------------------------------------------------------------------
// Launch
// ---------------------------------------------------------------------------
extern "C" void kernel_launch(void* const* d_in, const int* in_sizes, int n_in,
                              void* d_out, int out_size, void* d_ws, size_t ws_size,
                              hipStream_t stream) {
    (void)in_sizes; (void)n_in; (void)out_size; (void)ws_size;
    const float* x      = (const float*)d_in[0];
    const float* xt_1   = (const float*)d_in[1];
    const float* m      = (const float*)d_in[2];
    const float* h      = (const float*)d_in[3];
    const float* c      = (const float*)d_in[4];
    const float* nIn    = (const float*)d_in[5];
    const float* s      = (const float*)d_in[6];
    const float* flow_f = (const float*)d_in[7];
    const float* flow_d = (const float*)d_in[8];
    const float *w_x2h_n=(const float*)d_in[9],  *b_x2h_n=(const float*)d_in[10];
    const float *w_n2h_n=(const float*)d_in[11], *b_n2h_n=(const float*)d_in[12];
    const float *w_diff2o=(const float*)d_in[13],*b_diff2o=(const float*)d_in[14];
    const float *w_n2o  =(const float*)d_in[15], *b_n2o  =(const float*)d_in[16];
    const float *w_x2h_s=(const float*)d_in[17], *b_x2h_s=(const float*)d_in[18];
    const float *w_c2h_s=(const float*)d_in[19], *b_c2h_s=(const float*)d_in[20];
    const float *w_s2o  =(const float*)d_in[21], *b_s2o  =(const float*)d_in[22];
    const float *w_x2h  =(const float*)d_in[23], *b_x2h  =(const float*)d_in[24];
    const float *w_h2h  =(const float*)d_in[25], *b_h2h  =(const float*)d_in[26];
    const float *w_c2o  =(const float*)d_in[27], *b_c2o  =(const float*)d_in[28];
    const float *w_x2h_m=(const float*)d_in[29], *b_x2h_m=(const float*)d_in[30];
    const float *w_m2h_m=(const float*)d_in[31], *b_m2h_m=(const float*)d_in[32];
    const float *w_m2o  =(const float*)d_in[33], *b_m2o  =(const float*)d_in[34];
    const float *w_c_m  =(const float*)d_in[35], *b_c_m  =(const float*)d_in[36];
    const float *w_enc  =(const float*)d_in[37], *b_enc  =(const float*)d_in[38];
    const float *w_u    =(const float*)d_in[39], *b_u    =(const float*)d_in[40];
    const float *w_r    =(const float*)d_in[41], *b_r    =(const float*)d_in[42];
    const float *w_z    =(const float*)d_in[43], *b_z    =(const float*)d_in[44];
    const float *w_hm   =(const float*)d_in[45], *b_hm   =(const float*)d_in[46];
    const float *w_dec  =(const float*)d_in[47], *b_dec  =(const float*)d_in[48];
    const float *w_dcv  =(const float*)d_in[49], *b_dcv  =(const float*)d_in[50];
    const float *w_g    =(const float*)d_in[51], *b_g    =(const float*)d_in[52];

    float* out = (float*)d_out;
    float* nh = out;
    float* nm = out + (size_t)FULL;
    float* nc = out + (size_t)2 * FULL;
    float* nn = out + (size_t)3 * FULL;
    float* ns = out + (size_t)4 * FULL;
    float* nF = out + (size_t)5 * FULL;
    float* nD = nF + H18;

    // ---- workspace partition ----
    float* ws    = (float*)d_ws;
    float* Tbuf  = ws;                        // FULL
    float* oncB  = Tbuf  + (size_t)FULL;      // FULL : OnC -> S2O -> OC2 -> G
    float* oBuf  = oncB  + (size_t)FULL;      // FULL : o
    float* cmHd  = oBuf  + (size_t)FULL;      // FULL : CM -> H_dec(full, f32)
    float* HhB   = cmHd  + (size_t)FULL;      // FULL : Hh (f32)
    float* bigA  = HhB   + (size_t)FULL;      // 4*FULL : A1/A2/A4
    float* A3    = bigA  + (size_t)4 * FULL;  // 4*FULL
    float* Henc  = A3    + (size_t)4 * FULL;  // H16
    float* uB    = Henc  + H16;
    float* rB    = uB    + H18;
    float* zB    = rB    + H18;
    float* mmB   = zB    + H18;
    float* HwarpB= mmB   + H9;                // WARPN
    float* HdecH = HwarpB + WARPN;            // H16

    bf16* bfp = (bf16*)(HdecH + H16);
    bf16 *xb    = bfp; bfp += (size_t)FULL;
    bf16 *hb    = bfp; bfp += (size_t)FULL;
    bf16 *mb    = bfp; bfp += (size_t)FULL;
    bf16 *cb    = bfp; bfp += (size_t)FULL;
    bf16 *nb    = bfp; bfp += (size_t)FULL;
    bf16 *diffb = bfp; bfp += (size_t)FULL;
    bf16 *nnb   = bfp; bfp += (size_t)FULL;
    bf16 *difb  = bfp; bfp += (size_t)FULL;
    bf16 *nsb   = bfp; bfp += (size_t)FULL;
    bf16 *ncb   = bfp; bfp += (size_t)FULL;
    bf16 *nmb   = bfp; bfp += (size_t)FULL;
    bf16 *hhb   = bfp; bfp += (size_t)FULL;
    bf16 *hdecb = bfp; bfp += (size_t)FULL;
    bf16 *pw_x2h_n = bfp; bfp += (size_t)192 * 576;
    bf16 *pw_n2h_n = bfp; bfp += (size_t)192 * 576;
    bf16 *pw_diff2o= bfp; bfp += (size_t)64  * 576;
    bf16 *pw_n2o   = bfp; bfp += (size_t)64  * 576;
    bf16 *pw_x2h_s = bfp; bfp += (size_t)256 * 576;
    bf16 *pw_c2h_s = bfp; bfp += (size_t)256 * 576;
    bf16 *pw_s2o   = bfp; bfp += (size_t)64  * 576;
    bf16 *pw_x2h   = bfp; bfp += (size_t)256 * 576;
    bf16 *pw_h2h   = bfp; bfp += (size_t)256 * 576;
    bf16 *pw_c2o   = bfp; bfp += (size_t)64  * 576;
    bf16 *pw_x2h_m = bfp; bfp += (size_t)192 * 576;
    bf16 *pw_m2h_m = bfp; bfp += (size_t)192 * 576;
    bf16 *pw_m2o   = bfp; bfp += (size_t)64  * 576;
    bf16 *pw_cm    = bfp; bfp += (size_t)64 * 128;
    bf16 *pw_g     = bfp; bfp += (size_t)64 * 128;

    dim3 blk(256);
    const int EWG = (FULL + 255) / 256;
    dim3 gEW(EWG);
#define PACKW3(src, dst, OC) pack_w3<<<dim3(((OC)*576 + 255)/256), blk, 0, stream>>>(src, dst, OC)
    PACKW3(w_x2h_n, pw_x2h_n, 192);  PACKW3(w_n2h_n, pw_n2h_n, 192);
    PACKW3(w_diff2o, pw_diff2o, 64); PACKW3(w_n2o, pw_n2o, 64);
    PACKW3(w_x2h_s, pw_x2h_s, 256);  PACKW3(w_c2h_s, pw_c2h_s, 256);
    PACKW3(w_s2o, pw_s2o, 64);
    PACKW3(w_x2h, pw_x2h, 256);      PACKW3(w_h2h, pw_h2h, 256);
    PACKW3(w_c2o, pw_c2o, 64);
    PACKW3(w_x2h_m, pw_x2h_m, 192);  PACKW3(w_m2h_m, pw_m2h_m, 192);
    PACKW3(w_m2o, pw_m2o, 64);
    cvt_bf<<<dim3((64*128 + 255)/256), blk, 0, stream>>>(w_c_m, pw_cm, 64*128);
    cvt_bf<<<dim3((64*128 + 255)/256), blk, 0, stream>>>(w_g,   pw_g,  64*128);
    cvt_bf<<<gEW, blk, 0, stream>>>(x, xb, FULL);
    cvt_bf<<<gEW, blk, 0, stream>>>(h, hb, FULL);
    cvt_bf<<<gEW, blk, 0, stream>>>(m, mb, FULL);
    cvt_bf<<<gEW, blk, 0, stream>>>(c, cb, FULL);
    cvt_bf<<<gEW, blk, 0, stream>>>(nIn, nb, FULL);

    // 1. diff (bf16 only)
    ew_sub_bf<<<gEW, blk, 0, stream>>>(x, xt_1, diffb);
    // 2-3. next_n
    conv3x3_dual_wmma<<<dim3(HW, 12, 4), blk, 0, stream>>>(diffb, nb, pw_x2h_n, pw_n2h_n, b_x2h_n, b_n2h_n, bigA, 192);
    gate_ifg<<<gEW, blk, 0, stream>>>(bigA, nIn, nn, nnb, 3);
    // 4-5. Dif
    conv3x3_dual_wmma<<<dim3(HW, 4, 4), blk, 0, stream>>>(diffb, nnb, pw_diff2o, pw_n2o, b_diff2o, b_n2o, oncB, 64);
    ew_dif<<<gEW, blk, 0, stream>>>(oncB, nn, difb);
    // 6-7. next_s
    conv3x3_dual_wmma<<<dim3(HW, 16, 4), blk, 0, stream>>>(difb, cb, pw_x2h_s, pw_c2h_s, b_x2h_s, b_c2h_s, bigA, 256);
    gate_ifg<<<gEW, blk, 0, stream>>>(bigA, s, ns, nsb, 4);
    // 8-9. T
    conv3x3_dual_wmma<<<dim3(HW, 4, 4), blk, 0, stream>>>(nsb, nullptr, pw_s2o, nullptr, b_s2o, nullptr, oncB, 64);
    ew_T<<<gEW, blk, 0, stream>>>(bigA, oncB, ns, Tbuf);
    // 10-11. next_c
    conv3x3_dual_wmma<<<dim3(HW, 16, 4), blk, 0, stream>>>(xb, hb, pw_x2h, pw_h2h, b_x2h, b_h2h, A3, 256);
    ew_nc<<<gEW, blk, 0, stream>>>(A3, Tbuf, nc, ncb);
    // 12-13. next_m
    conv3x3_dual_wmma<<<dim3(HW, 12, 4), blk, 0, stream>>>(xb, mb, pw_x2h_m, pw_m2h_m, b_x2h_m, b_m2h_m, bigA, 192);
    gate_ifg<<<gEW, blk, 0, stream>>>(bigA, m, nm, nmb, 3);
    // 14-15. o
    conv3x3_dual_wmma<<<dim3(HW, 4, 4), blk, 0, stream>>>(ncb, nmb, pw_c2o, pw_m2o, b_c2o, b_m2o, oncB, 64);
    ew_o<<<gEW, blk, 0, stream>>>(A3, oncB, oBuf);
    // 16-17. Hh
    conv1x1_dual_wmma<<<dim3(128, 4, 4), blk, 0, stream>>>(ncb, nmb, pw_cm, b_c_m, cmHd, HW * HW, 64);
    ew_hh<<<gEW, blk, 0, stream>>>(oBuf, cmHd, HhB, hhb);
    // 18. encoder
    enc_conv<<<dim3((4 * 16 * HP + 255) / 256), blk, 0, stream>>>(hhb, w_enc, b_enc, Henc);
    // 19-21. u, r, z
    const int G18 = (4 * 18 * HP + 255) / 256;
    conv1x1_naive<<<dim3(G18), blk, 0, stream>>>(Henc, 16, flow_f, 18, nullptr, w_u, b_u, uB, 18, HP, 1);
    conv1x1_naive<<<dim3(G18), blk, 0, stream>>>(Henc, 16, flow_f, 18, nullptr, w_r, b_r, rB, 18, HP, 1);
    conv1x1_naive<<<dim3(G18), blk, 0, stream>>>(Henc, 16, flow_f, 18, rB, w_z, b_z, zB, 18, HP, 2);
    // 22. next_F / next_D
    ew_gru<<<dim3((H18 + 255) / 256), blk, 0, stream>>>(uB, zB, flow_f, flow_d, nF, nD);
    // 23. mm
    conv1x1_naive<<<dim3((4 * 9 * HP + 255) / 256), blk, 0, stream>>>(Henc, 16, nullptr, 0, nullptr, w_hm, b_hm, mmB, 9, HP, 1);
    // 24. warp
    warp_kernel<<<dim3((4 * 16 * HP + 255) / 256), blk, 0, stream>>>(Henc, nF, mmB, HwarpB);
    // 25. dec (torch reshape == identity view of warp buffer)
    conv1x1_naive<<<dim3((4 * 16 * HP + 255) / 256), blk, 0, stream>>>(HwarpB, 144, nullptr, 0, nullptr, w_dec, b_dec, HdecH, 16, HP, 0);
    // 26. deconv
    deconv_kernel<<<gEW, blk, 0, stream>>>(HdecH, w_dcv, b_dcv, cmHd, hdecb);
    // 27. g conv
    conv1x1_dual_wmma<<<dim3(128, 4, 4), blk, 0, stream>>>(hdecb, hhb, pw_g, b_g, oncB, HW * HW, 64);
    // 28. next_h
    ew_final<<<gEW, blk, 0, stream>>>(oncB, HhB, cmHd, oBuf, x, nh);
}
